// Network_62345745269444
// MI455X (gfx1250) — compile-verified, matrix operand order checked by reading, not run on hardware
//
#include <hip/hip_runtime.h>
#include <hip/hip_bf16.h>

typedef __attribute__((ext_vector_type(16))) _Float16 v16h;
typedef __attribute__((ext_vector_type(8)))  _Float16 v8h;
typedef __attribute__((ext_vector_type(8)))  float    v8f;

#define TPB 256
#define BATCH_N 128
#define HW0 32

// ---------------------------------------------------------------------------
// Pack weights [M][K] f32 -> [Mpad][Kpad] f16, zero padded.
// ---------------------------------------------------------------------------
static __global__ void k_pack_w(const float* __restrict__ W, _Float16* __restrict__ Wp,
                                int M, int K, int Kpad, long total) {
  long i = (long)blockIdx.x * blockDim.x + threadIdx.x;
  if (i >= total) return;
  int k = (int)(i % Kpad);
  int m = (int)(i / Kpad);
  float v = (m < M && k < K) ? W[(long)m * K + k] : 0.f;
  Wp[i] = (_Float16)v;
}

// ---------------------------------------------------------------------------
// Pack activations NCHW f32 [B][K][P2] -> pixel-major f16 [B*P2][Kpad]
// (k contiguous per pixel; optional fused ReLU). Coalesced reads.
// ---------------------------------------------------------------------------
static __global__ void k_pack_x(const float* __restrict__ X, _Float16* __restrict__ Xp,
                                int K, int P2, int Kpad, long total, int relu) {
  long i = (long)blockIdx.x * blockDim.x + threadIdx.x;   // over B*K*P2
  if (i >= total) return;
  int p = (int)(i % P2); long t = i / P2;
  int k = (int)(t % K);  int img = (int)(t / K);
  float v = X[i];
  if (relu) v = fmaxf(v, 0.f);
  Xp[((long)img * P2 + p) * Kpad + k] = (_Float16)v;
}

// zero only the pad columns K..Kpad-1
static __global__ void k_pad_x(_Float16* __restrict__ Xp, int K, int P2, int Kpad, long total) {
  long i = (long)blockIdx.x * blockDim.x + threadIdx.x;   // over B*P2*(Kpad-K)
  if (i >= total) return;
  int pad = Kpad - K;
  int kk = K + (int)(i % pad); long t = i / pad;
  int p = (int)(t % P2); int img = (int)(t / P2);
  Xp[((long)img * P2 + p) * Kpad + kk] = (_Float16)0.f;
}

// ---------------------------------------------------------------------------
// Per-lane 16-half fragment: two aligned 16B runs (k = hi*8+0..7, 16+hi*8+0..7)
// per CDNA5 ISA 7.12.2 16-bit A/B fragment layout.
// ---------------------------------------------------------------------------
__device__ __forceinline__ v16h ldfrag(const _Float16* p) {
  v8h lo = *(const v8h*)p;
  v8h hi = *(const v8h*)(p + 16);
  return __builtin_shufflevector(lo, hi, 0,1,2,3,4,5,6,7,8,9,10,11,12,13,14,15);
}

// ---------------------------------------------------------------------------
// WMMA GEMM for 1x1 conv.
//   Wp : Mpad x Kpad   f16 (zero padded)
//   Xp : (B*P2) x Kpad f16 (pixel-major, zero padded)
//   Y  : B x M x P2    f32 (NCHW)
// Block = 4 waves; each wave computes a 32(M) x 32(pixels) tile with 4 WMMAs
// per 32-k chunk (each A/B fragment feeds two WMMAs -> 2x operand reuse).
// No predication inside the k-loop; EXEC all-ones at every WMMA.
// ---------------------------------------------------------------------------
static __global__ __launch_bounds__(128)
void k_gemm_wmma(const _Float16* __restrict__ Wp, const _Float16* __restrict__ Xp,
                 float* __restrict__ Y, int M, int Kpad, int P2) {
  const int lane = threadIdx.x;                       // 0..31 (wave32)
  const int col  = lane & 15;
  const int hi   = lane >> 4;                         // 0|1
  const long pixT = (long)blockIdx.x * 4 + threadIdx.y;
  const long pix0 = pixT * 32 + col;                  // first 16-pixel tile
  const long pix1 = pix0 + 16;                        // second 16-pixel tile
  const int img0 = (int)(pix0 / P2), p0 = (int)(pix0 % P2);
  const int img1 = (int)(pix1 / P2), p1 = (int)(pix1 % P2);
  const int m0   = blockIdx.y * 32;

  const _Float16* brow0 = Xp + pix0 * Kpad + hi * 8;
  const _Float16* brow1 = Xp + pix1 * Kpad + hi * 8;
  const _Float16* arow0 = Wp + (long)(m0 + col) * Kpad + hi * 8;
  const _Float16* arow1 = arow0 + (long)16 * Kpad;

  v8f acc00 = {}, acc01 = {}, acc10 = {}, acc11 = {};
  for (int k0 = 0; k0 < Kpad; k0 += 32) {
    __builtin_prefetch((const void*)(brow0 + k0 + 128), 0, 1);
    __builtin_prefetch((const void*)(brow1 + k0 + 128), 0, 1);
    v16h a0 = ldfrag(arow0 + k0);
    v16h a1 = ldfrag(arow1 + k0);
    v16h b0 = ldfrag(brow0 + k0);
    v16h b1 = ldfrag(brow1 + k0);
    acc00 = __builtin_amdgcn_wmma_f32_16x16x32_f16(false, a0, false, b0, (short)0, acc00, false, false);
    acc10 = __builtin_amdgcn_wmma_f32_16x16x32_f16(false, a1, false, b0, (short)0, acc10, false, false);
    acc01 = __builtin_amdgcn_wmma_f32_16x16x32_f16(false, a0, false, b1, (short)0, acc01, false, false);
    acc11 = __builtin_amdgcn_wmma_f32_16x16x32_f16(false, a1, false, b1, (short)0, acc11, false, false);
  }
#pragma unroll
  for (int r = 0; r < 8; ++r) {
    int mm = m0 + r + 8 * hi;                        // C/D layout: vgpr r, half hi
    if (mm < M) {
      Y[((long)img0 * M + mm) * P2 + p0] = acc00[r];
      Y[((long)img1 * M + mm) * P2 + p1] = acc01[r];
    }
    if (mm + 16 < M) {
      Y[((long)img0 * M + mm + 16) * P2 + p0] = acc10[r];
      Y[((long)img1 * M + mm + 16) * P2 + p1] = acc11[r];
    }
  }
}

// ---------------------------------------------------------------------------
// Stem 3x3 conv, stride 1, pad 1 (Cin=3 -> tiny K, plain VALU)
// ---------------------------------------------------------------------------
static __global__ void k_conv3x3_stem(const float* __restrict__ x, const float* __restrict__ w,
                                      float* __restrict__ y, int B, int Cin, int Cout, int H) {
  long total = (long)B * Cout * H * H;
  long i = (long)blockIdx.x * blockDim.x + threadIdx.x;
  if (i >= total) return;
  int wp = i % H; long t = i / H;
  int hp = t % H; t /= H;
  int co = t % Cout; int n = (int)(t / Cout);
  float acc = 0.f;
  for (int ci = 0; ci < Cin; ++ci) {
    const float* xp = x + ((long)n * Cin + ci) * H * H;
    const float* wq = w + ((long)co * Cin + ci) * 9;
#pragma unroll
    for (int kh = 0; kh < 3; ++kh) {
      int ih = hp + kh - 1;
      if (ih < 0 || ih >= H) continue;
#pragma unroll
      for (int kw = 0; kw < 3; ++kw) {
        int iw = wp + kw - 1;
        if (iw < 0 || iw >= H) continue;
        acc += xp[ih * H + iw] * wq[kh * 3 + kw];
      }
    }
  }
  y[i] = acc;
}

// ---------------------------------------------------------------------------
// Depthwise 3x3 conv, pad 1, stride 1|2, ReLU fused on input.  w: [C][3][3]
// ---------------------------------------------------------------------------
static __global__ void k_dwconv3x3(const float* __restrict__ x, const float* __restrict__ w,
                                   float* __restrict__ y, int B, int C, int H, int stride) {
  int Ho = H / stride;
  long total = (long)B * C * Ho * Ho;
  long i = (long)blockIdx.x * blockDim.x + threadIdx.x;
  if (i >= total) return;
  int ow = i % Ho; long t = i / Ho;
  int oh = t % Ho; t /= Ho;
  int c = t % C; int n = (int)(t / C);
  const float* xp = x + ((long)n * C + c) * H * H;
  const float* wq = w + c * 9;
  float acc = 0.f;
#pragma unroll
  for (int kh = 0; kh < 3; ++kh) {
    int ih = oh * stride + kh - 1;
    if (ih < 0 || ih >= H) continue;
#pragma unroll
    for (int kw = 0; kw < 3; ++kw) {
      int iw = ow * stride + kw - 1;
      if (iw < 0 || iw >= H) continue;
      acc += fmaxf(xp[ih * H + iw], 0.f) * wq[kh * 3 + kw];
    }
  }
  y[i] = acc;
}

// ---------------------------------------------------------------------------
// BatchNorm stats: one block per channel (mean over B*P2, population var)
// ---------------------------------------------------------------------------
static __global__ void k_bn_stats(const float* __restrict__ x, float* __restrict__ mean,
                                  float* __restrict__ rstd, int B, int C, int P2) {
  int c = blockIdx.x;
  long n = (long)B * P2;
  float s = 0.f, s2 = 0.f;
  for (long i = threadIdx.x; i < n; i += blockDim.x) {
    long img = i / P2, p = i % P2;
    float v = x[(img * C + c) * P2 + p];
    s += v; s2 += v * v;
  }
  __shared__ float sh[256], sh2[256];
  sh[threadIdx.x] = s; sh2[threadIdx.x] = s2;
  __syncthreads();
  for (int st = 128; st > 0; st >>= 1) {
    if (threadIdx.x < st) { sh[threadIdx.x] += sh[threadIdx.x + st]; sh2[threadIdx.x] += sh2[threadIdx.x + st]; }
    __syncthreads();
  }
  if (threadIdx.x == 0) {
    float m = sh[0] / (float)n;
    float v = sh2[0] / (float)n - m * m;
    mean[c] = m;
    rstd[c] = rsqrtf(v + 1e-5f);
  }
}

static __global__ void k_bn_apply(const float* __restrict__ x, float* __restrict__ y,
                                  const float* __restrict__ mean, const float* __restrict__ rstd,
                                  int C, int P2, long total) {
  long i = (long)blockIdx.x * blockDim.x + threadIdx.x;
  if (i >= total) return;
  int c = (int)((i / P2) % C);
  y[i] = (x[i] - mean[c]) * rstd[c];
}

// ---------------------------------------------------------------------------
// Factorized reduce (BN applied by host wrapper afterwards).
// ---------------------------------------------------------------------------
static __global__ void k_fact_reduce(const float* __restrict__ x, const float* __restrict__ w1,
                                     const float* __restrict__ w2, float* __restrict__ y,
                                     int B, int Cin, int Cout, int H) {
  int Ho = H / 2;
  long total = (long)B * Cout * Ho * Ho;
  long i = (long)blockIdx.x * blockDim.x + threadIdx.x;
  if (i >= total) return;
  int ow = i % Ho; long t = i / Ho;
  int oh = t % Ho; t /= Ho;
  int co = t % Cout; int n = (int)(t / Cout);
  int Chalf = Cout / 2;
  int sh = (co < Chalf) ? 0 : 1;
  const float* wsel = (co < Chalf) ? (w1 + (long)co * Cin) : (w2 + (long)(co - Chalf) * Cin);
  int ih = 2 * oh + sh, iw = 2 * ow + sh;
  const float* xp = x + ((long)n * Cin) * H * H + ih * H + iw;
  float acc = 0.f;
  for (int ci = 0; ci < Cin; ++ci)
    acc += fmaxf(xp[(long)ci * H * H], 0.f) * wsel[ci];
  y[i] = acc;
}

// ---------------------------------------------------------------------------
// Mixed-op combine:  dst (+)= gw0*skip + gw1*sep
// ---------------------------------------------------------------------------
static __global__ void k_combine(const float* __restrict__ skip, const float* __restrict__ sep,
                                 float* __restrict__ dst, const float* __restrict__ gw,
                                 int gwidx, long n, int first) {
  long i = (long)blockIdx.x * blockDim.x + threadIdx.x;
  if (i >= n) return;
  float a = gw[gwidx * 2 + 0];
  float b = gw[gwidx * 2 + 1];
  float v = a * skip[i] + b * sep[i];
  dst[i] = first ? v : (dst[i] + v);
}

// ---------------------------------------------------------------------------
// Concat 4 node states [B][C][P2] -> [B][4C][P2]
// ---------------------------------------------------------------------------
static __global__ void k_concat4(const float* __restrict__ s2, const float* __restrict__ s3,
                                 const float* __restrict__ s4, const float* __restrict__ s5,
                                 float* __restrict__ out, int B, int C, int P2) {
  long total = (long)B * 4 * C * P2;
  long i = (long)blockIdx.x * blockDim.x + threadIdx.x;
  if (i >= total) return;
  int p = i % P2; long t = i / P2;
  int c = t % (4 * C); int n = (int)(t / (4 * C));
  int which = c / C, cc = c % C;
  const float* s = (which == 0) ? s2 : (which == 1) ? s3 : (which == 2) ? s4 : s5;
  out[i] = s[((long)n * C + cc) * P2 + p];
}

// ---------------------------------------------------------------------------
// Global average pool + classifier
// ---------------------------------------------------------------------------
static __global__ void k_pool(const float* __restrict__ x, float* __restrict__ y,
                              int BC, int P2) {
  int i = blockIdx.x * blockDim.x + threadIdx.x;
  if (i >= BC) return;
  const float* p = x + (long)i * P2;
  float s = 0.f;
  for (int k = 0; k < P2; ++k) s += p[k];
  y[i] = s / (float)P2;
}

static __global__ void k_cls(const float* __restrict__ pooled, const float* __restrict__ W,
                             const float* __restrict__ bvec, float* __restrict__ out,
                             int B, int C, int NC) {
  int i = blockIdx.x * blockDim.x + threadIdx.x;
  if (i >= B * NC) return;
  int n = i / NC, c = i % NC;
  float s = bvec[c];
  const float* pr = pooled + (long)n * C;
  const float* wr = W + (long)c * C;
  for (int k = 0; k < C; ++k) s += pr[k] * wr[k];
  out[i] = s;
}

// ---------------------------------------------------------------------------
// Edge weights (gated sigmoid(alphas)) + FLOPs regularizer scalar.
// ---------------------------------------------------------------------------
static __global__ void k_weights_flops(const float* __restrict__ alphas,
                                       float* __restrict__ gw, float* __restrict__ flops_out) {
  __shared__ float w[8][14][2];
  __shared__ float nodesum[8][4];
  const int bs[5] = {0, 2, 5, 9, 14};
  int t = threadIdx.x;
  if (t < 112) {
    int cell = t / 14, e = t % 14;
    float a0 = alphas[(cell * 14 + e) * 2 + 0];
    float a1 = alphas[(cell * 14 + e) * 2 + 1];
    w[cell][e][0] = 1.f / (1.f + expf(-a0));
    w[cell][e][1] = 1.f / (1.f + expf(-a1));
  }
  __syncthreads();
  if (t < 32) {
    int cell = t / 4, blk = t % 4;
    float s = 0.f;
    for (int e = bs[blk]; e < bs[blk + 1]; ++e) s += w[cell][e][0] + w[cell][e][1];
    nodesum[cell][blk] = s;
  }
  __syncthreads();
  if (t < 112) {
    int cell = t / 14, e = t % 14;
    int blk = (e < 2) ? 0 : (e < 5) ? 1 : (e < 9) ? 2 : 3;
    float ns = nodesum[cell][blk];
    float w0 = w[cell][e][0], w1 = w[cell][e][1];
    gw[(cell * 14 + e) * 2 + 0] = (w0 > 0.01f * ns) ? w0 : 0.f;
    gw[(cell * 14 + e) * 2 + 1] = (w1 > 0.01f * ns) ? w1 : 0.f;
  }
  __syncthreads();
  if (t == 0) {
    const int redlist[14] = {1, 1, 1, 1, 0, 1, 1, 0, 0, 1, 1, 0, 0, 0};
    float fl = 0.f;
    for (int cell = 0; cell < 8; ++cell) {
      int red = (cell == 2 || cell == 5);
      float Cc = (cell < 2) ? 36.f : (cell < 5) ? 72.f : 144.f;
      for (int e = 0; e < 14; ++e) {
        int blk = (e < 2) ? 0 : (e < 5) ? 1 : (e < 9) ? 2 : 3;
        float ns = nodesum[cell][blk];
        float ops = 0.f;
        for (int ee = bs[blk]; ee < bs[blk + 1]; ++ee) {
          if (w[cell][ee][0] / ns > 0.01f) ops += 1.f;
          if (w[cell][ee][1] / ns > 0.01f) ops += 1.f;
        }
        int skip_red = redlist[e] && red;
        for (int op = 0; op < 2; ++op) {
          float wv = w[cell][e][op];
          float fc;
          if (op == 1)
            fc = 1e-6f * 2.f * (1296.f * 1024.f + 9.f * 1024.f * 1296.f / Cc);
          else if (skip_red)
            fc = 1e-6f * 1296.f * 1024.f;
          else
            fc = 0.f;
          float coef = 0.01f + 1.f * fc;
          float r = wv / ns;
          if (r > 0.01f) fl += log1pf(ops * r) * coef;
        }
      }
    }
    flops_out[0] = fl;
  }
}

// ---------------------------------------------------------------------------
// Host orchestration
// ---------------------------------------------------------------------------
extern "C" void kernel_launch(void* const* d_in, const int* in_sizes, int n_in,
                              void* d_out, int out_size, void* d_ws, size_t ws_size,
                              hipStream_t stream) {
  (void)in_sizes; (void)n_in; (void)out_size; (void)ws_size;

  const int redlist[14] = {1, 1, 1, 1, 0, 1, 1, 0, 0, 1, 1, 0, 0, 0};

  struct Spec { int Cpp, Cp, C; bool red, red_prev; };
  Spec specs[8];
  {
    int Cpp = 108, Cp = 108, Cc = 36; bool rp = false;
    for (int i = 0; i < 8; ++i) {
      bool red = (i == 2 || i == 5);
      if (red) Cc *= 2;
      specs[i] = {Cpp, Cp, Cc, red, rp};
      rp = red;
      int nCp = 4 * Cc;
      Cpp = Cp; Cp = nCp;
    }
  }

  // ---- parameter walker (insertion order of the reference pytree) ----
  int pi = 0;
  auto nextp = [&]() { return (const float*)d_in[pi++]; };
  const float* xin    = nextp();
  const float* alphas = nextp();
  const float* stem_w = nextp();
  struct Cell {
    const float *pre0w, *pre0w1, *pre0w2, *pre1w;
    const float *dw1[14], *pw1[14], *dw2[14], *pw2[14], *fr1[14], *fr2[14];
  };
  Cell cells[8];
  for (int i = 0; i < 8; ++i) {
    Cell& c = cells[i];
    c.pre0w = c.pre0w1 = c.pre0w2 = nullptr;
    if (specs[i].red_prev) { c.pre0w1 = nextp(); c.pre0w2 = nextp(); }
    else                   { c.pre0w  = nextp(); }
    c.pre1w = nextp();
    for (int e = 0; e < 14; ++e) {
      c.dw1[e] = nextp(); c.pw1[e] = nextp();
      c.dw2[e] = nextp(); c.pw2[e] = nextp();
      c.fr1[e] = c.fr2[e] = nullptr;
      if (specs[i].red && redlist[e]) { c.fr1[e] = nextp(); c.fr2[e] = nextp(); }
    }
  }
  const float* cls_w = nextp();
  const float* cls_b = nextp();

  // ---- workspace layout (floats) ----
  const size_t OUTBUF  = 18874368;   // 128*144*1024 (max cell output / stem)
  const size_t PREBUF  = 9437184;    // 128*72*1024  (max preprocessed state)
  const size_t NODEBUF = 4718592;    // 128*36*1024  (max node state / edge temp)
  const size_t XPBUF   = 10485760;   // 20.97M halves: 128*1024*Kpad(160)
  float* A = (float*)d_ws;
  size_t off = 0;
  auto alloc = [&](size_t n) { float* p = A + off; off += n; return p; };
  float* ring[3] = {alloc(OUTBUF), alloc(OUTBUF), alloc(OUTBUF)};
  float* s0p  = alloc(PREBUF);
  float* s1p  = alloc(PREBUF);
  float* stN[4] = {alloc(NODEBUF), alloc(NODEBUF), alloc(NODEBUF), alloc(NODEBUF)};
  float* t1   = alloc(NODEBUF);
  float* t2   = alloc(NODEBUF);
  float* sepb = alloc(NODEBUF);
  float* skpb = alloc(NODEBUF);
  _Float16* xh = (_Float16*)alloc(XPBUF);    // packed activations
  _Float16* wh = (_Float16*)alloc(65536);    // packed weights (<=160*576 halves)
  float* meanb = alloc(1024);
  float* rstdb = alloc(1024);
  float* gwbuf = alloc(256);
  float* pooled = alloc(BATCH_N * 576);

  auto grid1 = [&](long n) { return dim3((unsigned)((n + TPB - 1) / TPB)); };

  auto bn = [&](float* x, int C, int P2) {
    long n = (long)BATCH_N * C * P2;
    k_bn_stats<<<C, 256, 0, stream>>>(x, meanb, rstdb, BATCH_N, C, P2);
    k_bn_apply<<<grid1(n), TPB, 0, stream>>>(x, x, meanb, rstdb, C, P2, n);
  };

  auto gemm = [&](const float* Wf, const float* Xf, float* Y,
                  int M, int K, int P2, bool reluX) {
    int Kpad = (K + 31) & ~31;
    int Mpad = (M + 31) & ~31;
    long nw = (long)Mpad * Kpad;
    k_pack_w<<<grid1(nw), TPB, 0, stream>>>(Wf, wh, M, K, Kpad, nw);
    long nx = (long)BATCH_N * K * P2;
    k_pack_x<<<grid1(nx), TPB, 0, stream>>>(Xf, xh, K, P2, Kpad, nx, reluX ? 1 : 0);
    if (Kpad > K) {
      long nz = (long)BATCH_N * P2 * (Kpad - K);
      k_pad_x<<<grid1(nz), TPB, 0, stream>>>(xh, K, P2, Kpad, nz);
    }
    dim3 g((unsigned)((BATCH_N * P2) / 128), (unsigned)(Mpad / 32));
    k_gemm_wmma<<<g, dim3(32, 4), 0, stream>>>(wh, xh, Y, M, Kpad, P2);
  };

  auto factred = [&](const float* x, const float* w1, const float* w2, float* y,
                     int Cin, int Cout, int H) {
    int Ho = H / 2;
    long n = (long)BATCH_N * Cout * Ho * Ho;
    k_fact_reduce<<<grid1(n), TPB, 0, stream>>>(x, w1, w2, y, BATCH_N, Cin, Cout, H);
    bn(y, Cout, Ho * Ho);
  };

  auto dwc = [&](const float* x, const float* w, float* y, int C, int H, int stride) {
    int Ho = H / stride;
    long n = (long)BATCH_N * C * Ho * Ho;
    k_dwconv3x3<<<grid1(n), TPB, 0, stream>>>(x, w, y, BATCH_N, C, H, stride);
  };

  auto sepconv = [&](const float* x, int C, int H, int stride,
                     const float* dw1, const float* pw1,
                     const float* dw2, const float* pw2, float* out) {
    int Ho = H / stride, P2 = Ho * Ho;
    dwc(x, dw1, t1, C, H, stride);
    gemm(pw1, t1, t2, C, C, P2, false);
    bn(t2, C, P2);
    dwc(t2, dw2, t1, C, Ho, 1);
    gemm(pw2, t1, out, C, C, P2, false);
    bn(out, C, P2);
  };

  // ---- edge weights + flops scalar (before any combine) ----
  k_weights_flops<<<1, 128, 0, stream>>>(alphas, gwbuf, ((float*)d_out) + BATCH_N * 10);

  // ---- stem ----
  {
    long n = (long)BATCH_N * 108 * HW0 * HW0;
    k_conv3x3_stem<<<grid1(n), TPB, 0, stream>>>(xin, stem_w, ring[0], BATCH_N, 3, 108, HW0);
    bn(ring[0], 108, HW0 * HW0);
  }

  // ---- cells ----
  const float* s0 = ring[0];
  const float* s1 = ring[0];
  int i0 = 0, i1 = 0;
  int H0 = HW0, H1 = HW0;
  for (int ci = 0; ci < 8; ++ci) {
    const Spec& sp = specs[ci];
    const Cell& cp = cells[ci];
    const int C = sp.C;

    if (sp.red_prev) {
      factred(s0, cp.pre0w1, cp.pre0w2, s0p, sp.Cpp, C, H0);
    } else {
      gemm(cp.pre0w, s0, s0p, C, sp.Cpp, H0 * H0, true);
      bn(s0p, C, H0 * H0);
    }
    gemm(cp.pre1w, s1, s1p, C, sp.Cp, H1 * H1, true);
    bn(s1p, C, H1 * H1);

    const int H = H1;
    const int Hout = sp.red ? H / 2 : H;
    const long nodeN = (long)BATCH_N * C * Hout * Hout;
    const float* st[6] = {s0p, s1p, stN[0], stN[1], stN[2], stN[3]};

    int eoff = 0;
    for (int step = 0; step < 4; ++step) {
      int ns = 2 + step;
      for (int j = 0; j < ns; ++j) {
        int e = eoff + j;
        bool str2 = sp.red && (j < 2);
        int Hin = str2 ? H : Hout;
        const float* xe = st[j];
        const float* skip;
        if (str2) {
          factred(xe, cp.fr1[e], cp.fr2[e], skpb, C, C, Hin);
          skip = skpb;
        } else {
          skip = xe;
        }
        sepconv(xe, C, Hin, str2 ? 2 : 1, cp.dw1[e], cp.pw1[e], cp.dw2[e], cp.pw2[e], sepb);
        k_combine<<<grid1(nodeN), TPB, 0, stream>>>(skip, sepb, stN[step], gwbuf,
                                                    ci * 14 + e, nodeN, (j == 0) ? 1 : 0);
      }
      eoff += ns;
    }

    int wr = 0;
    for (int s = 0; s < 3; ++s) if (s != i0 && s != i1) { wr = s; break; }
    float* outbuf = ring[wr];
    k_concat4<<<grid1(4 * nodeN), TPB, 0, stream>>>(stN[0], stN[1], stN[2], stN[3],
                                                    outbuf, BATCH_N, C, Hout * Hout);
    s0 = s1; i0 = i1;
    s1 = outbuf; i1 = wr;
    H0 = H1; H1 = Hout;
  }

  // ---- pool + classifier ----
  const int Cf = 576, P2f = H1 * H1;
  k_pool<<<grid1(BATCH_N * Cf), TPB, 0, stream>>>(s1, pooled, BATCH_N * Cf, P2f);
  k_cls<<<grid1(BATCH_N * 10), TPB, 0, stream>>>(pooled, cls_w, cls_b, (float*)d_out,
                                                 BATCH_N, Cf, 10);
}
